// AttentionRNN_523986010636
// MI455X (gfx1250) — matrix operand order repaired and checked by reading, not compiled, not run on hardware
//
#include <hip/hip_runtime.h>
#include <hip/hip_bf16.h>

typedef __attribute__((ext_vector_type(16))) __bf16 v16bf;
typedef __attribute__((ext_vector_type(8)))  __bf16 v8bf;
typedef __attribute__((ext_vector_type(8)))  float  v8f;
typedef __attribute__((ext_vector_type(4)))  float  v4f;

#define N_B 64
#define T_S 512
#define D_IN 128
#define H_SZ 256
#define H2 512
#define G4 1024

__device__ __forceinline__ v8f wmma_bf16(const v16bf& a, const v16bf& b, const v8f& c) {
  return __builtin_amdgcn_wmma_f32_16x16x32_bf16(false, a, false, b, (short)0, c, false, false);
}

__device__ __forceinline__ float sigm(float x) { return 1.0f / (1.0f + __expf(-x)); }

// Load 8 contiguous fp32 (16B-aligned) and place bf16-converted into v[base..base+7].
__device__ __forceinline__ void load8bf(const float* __restrict__ p, v16bf& v, int base) {
  const v4f a = *(const v4f*)p;
  const v4f b = *(const v4f*)(p + 4);
#pragma unroll
  for (int i = 0; i < 4; ++i) {
    v[base + i]     = (__bf16)a[i];
    v[base + 4 + i] = (__bf16)b[i];
  }
}

// ---------------------------------------------------------------------------
// WMMA GEMM: C[M x wcols(ldc)] = act(A[M x K] @ W[Nout x K]^T + bias)
// Guard-free hot loop: K % 32 == 0, A rows 16B-aligned (K % 8 == 0), W rows
// addressed with clamped bcol so pad lanes load in-range garbage; pad columns
// (Nout <= col < wcols) are stored as literal 0 so downstream layers can use
// the padded buffer as a K%32 input. One 16x16 tile per wave, 4 waves/block.
// ---------------------------------------------------------------------------
__global__ __launch_bounds__(128)
void gemm_wmma(const float* __restrict__ A, const float* __restrict__ W,
               const float* __restrict__ bias, float* __restrict__ C,
               int M, int K, int Nout, int ldc, int wcols, int relu)
{
  const int lane = threadIdx.x & 31;
  const int wave = threadIdx.x >> 5;
  const int ntiles = (wcols + 15) >> 4;
  const int mtiles = M >> 4;
  const int tile = blockIdx.x * 4 + wave;
  if (tile >= mtiles * ntiles) return;
  const int tm = tile / ntiles;
  const int tn = tile % ntiles;

  const int l15 = lane & 15;
  const int lhi = lane >> 4;
  const int arow  = tm * 16 + l15;
  const int bcol0 = tn * 16 + l15;
  const int bcol  = min(bcol0, Nout - 1);  // clamp: loads always in-range
  const float* __restrict__ Ap = A + (long)arow * K;
  const float* __restrict__ Wp = W + (long)bcol * K;

  v8f acc = {};
  for (int kk = 0; kk < K; kk += 32) {
    v16bf a, b;
    const int ka = kk + lhi * 16;
    load8bf(Ap + ka,      a, 0);
    load8bf(Ap + ka + 8,  a, 8);
    const int kb = kk + lhi * 16;
    load8bf(Wp + kb,     b, 0);
    load8bf(Wp + kb + 8, b, 8);
    acc = wmma_bf16(a, b, acc);
  }

  if (bcol0 >= wcols) return;               // nothing to write (hit when wcols % 16 != 0, e.g. wcols=2)
  const bool valid = (bcol0 < Nout);
  const float bv = valid ? bias[bcol0] : 0.0f;
#pragma unroll
  for (int r = 0; r < 8; ++r) {
    const int row = tm * 16 + r + lhi * 8;
    float v = acc[r] + bv;
    if (relu) v = fmaxf(v, 0.0f);
    if (!valid) v = 0.0f;                   // zero-fill pad columns
    C[(long)row * ldc + bcol0] = v;
  }
}

// ---------------------------------------------------------------------------
// Zero-padded weight repack for the small FC layers: Wp[Nout x Kpad].
// ---------------------------------------------------------------------------
__global__ __launch_bounds__(256)
void pack_w(const float* __restrict__ W, float* __restrict__ Wp,
            int Nout, int K, int Kpad)
{
  const int idx = blockIdx.x * 256 + threadIdx.x;
  if (idx >= Nout * Kpad) return;
  const int n = idx / Kpad, k = idx % Kpad;
  Wp[idx] = (k < K) ? W[n * K + k] : 0.0f;
}

// ---------------------------------------------------------------------------
// Persistent per-direction LSTM. grid = 2 (blockIdx.x: 0=fwd, 1=bwd).
// 1024 threads = 32 waves. Wave w: hidden cols [(w>>1)*16, +16) for ALL 4
// gates, batch-tile pair (w&1). h kept in LDS (bf16), c in registers.
// Per step: 64 WMMAs/wave. gin = precomputed x@W_ih^T + b, row = n*T + t.
// ---------------------------------------------------------------------------
__global__ __launch_bounds__(1024)
void lstm_dir(const float* __restrict__ gin_f, const float* __restrict__ gin_b,
              const float* __restrict__ whh_f, const float* __restrict__ whh_b,
              float* __restrict__ hs)
{
  const int dir = blockIdx.x;
  const float* __restrict__ gin = dir ? gin_b : gin_f;
  const float* __restrict__ whh = dir ? whh_b : whh_f;

  __shared__ __align__(16) __bf16 hsh[N_B * H_SZ];  // 32 KB

  const int tid  = threadIdx.x;
  const int lane = tid & 31;
  const int wave = tid >> 5;
  const int l15  = lane & 15;
  const int lhi  = lane >> 4;
  const int hbase = (wave >> 1) * 16;
  const int mbase = (wave & 1) * 32;

  for (int i = tid; i < N_B * H_SZ; i += 1024) hsh[i] = (__bf16)0.0f;

  float creg[2][8];
#pragma unroll
  for (int mt = 0; mt < 2; ++mt)
#pragma unroll
    for (int r = 0; r < 8; ++r) creg[mt][r] = 0.0f;

  __syncthreads();

  for (int t = 0; t < T_S; ++t) {
    const int tt = dir ? (T_S - 1 - t) : t;

    v8f acc[2][4];
#pragma unroll
    for (int mt = 0; mt < 2; ++mt)
#pragma unroll
      for (int g = 0; g < 4; ++g) {
#pragma unroll
        for (int r = 0; r < 8; ++r) {
          const int n   = mbase + mt * 16 + r + lhi * 8;
          const int col = g * H_SZ + hbase + l15;
          acc[mt][g][r] = gin[((long)n * T_S + tt) * G4 + col];
        }
      }

    for (int kk = 0; kk < H_SZ; kk += 32) {
      v16bf a[2];
#pragma unroll
      for (int mt = 0; mt < 2; ++mt) {
        const int m  = mbase + mt * 16 + l15;
        const int ka = kk + lhi * 16;
        const v8bf lo = *(const v8bf*)&hsh[m * H_SZ + ka];
        const v8bf hi = *(const v8bf*)&hsh[m * H_SZ + ka + 8];
#pragma unroll
        for (int i = 0; i < 8; ++i) { a[mt][i] = lo[i]; a[mt][i + 8] = hi[i]; }
      }
      v16bf b[4];
#pragma unroll
      for (int g = 0; g < 4; ++g) {
        const int ncol = g * H_SZ + hbase + l15;
        const float* wrow = whh + (long)ncol * H_SZ + kk + lhi * 16;
        load8bf(wrow,     b[g], 0);
        load8bf(wrow + 8, b[g], 8);
      }
#pragma unroll
      for (int mt = 0; mt < 2; ++mt)
#pragma unroll
        for (int g = 0; g < 4; ++g)
          acc[mt][g] = wmma_bf16(a[mt], b[g], acc[mt][g]);
    }

    __syncthreads();  // all waves done reading hsh(t-1)

#pragma unroll
    for (int mt = 0; mt < 2; ++mt) {
#pragma unroll
      for (int r = 0; r < 8; ++r) {
        const float ig = sigm(acc[mt][0][r]);
        const float fg = sigm(acc[mt][1][r]);
        const float gg = tanhf(acc[mt][2][r]);
        const float og = sigm(acc[mt][3][r]);
        const float cn = fg * creg[mt][r] + ig * gg;
        creg[mt][r] = cn;
        const float h = og * tanhf(cn);
        const int n = mbase + mt * 16 + r + lhi * 8;
        hsh[n * H_SZ + hbase + l15] = (__bf16)h;
        hs[((long)n * T_S + tt) * H2 + dir * H_SZ + hbase + l15] = h;
      }
    }
    __syncthreads();  // hsh(t) visible to all waves
  }
}

// ---------------------------------------------------------------------------
// Attention: per (n, 16-row query block). S = hs hs^T (WMMA), LDS softmax,
// ctx = P hs (WMMA). Writes [ctx | hs] into out4h (N,T,1024).
// ---------------------------------------------------------------------------
__global__ __launch_bounds__(256)
void attn_kernel(const float* __restrict__ hs, float* __restrict__ out4h)
{
  const int n  = blockIdx.y;
  const int qb = blockIdx.x * 16;
  const float* __restrict__ hsn = hs + (long)n * T_S * H2;

  __shared__ __align__(16) float  S[16 * T_S];   // 32 KB
  __shared__ __align__(16) __bf16 P[16 * T_S];   // 16 KB
  __shared__ float red[16][16];
  __shared__ float rowstat[16];

  const int tid = threadIdx.x, lane = tid & 31, wave = tid >> 5;
  const int l15 = lane & 15, lhi = lane >> 4;

  // phase 1: scores (both operands contiguous per lane -> b128 loads)
  for (int ct = wave * 4; ct < wave * 4 + 4; ++ct) {
    v8f acc = {};
    const float* qrow = hsn + (long)(qb + l15) * H2;
    const float* krow = hsn + (long)(ct * 16 + l15) * H2;
    for (int kk = 0; kk < H2; kk += 32) {
      v16bf a, b;
      const int ka = kk + lhi * 16;
      load8bf(qrow + ka,      a, 0);
      load8bf(qrow + ka + 8,  a, 8);
      const int kb = kk + lhi * 16;
      load8bf(krow + kb,     b, 0);
      load8bf(krow + kb + 8, b, 8);
      acc = wmma_bf16(a, b, acc);
    }
#pragma unroll
    for (int r = 0; r < 8; ++r)
      S[(r + lhi * 8) * T_S + ct * 16 + l15] = acc[r];
  }
  __syncthreads();

  // phase 2: softmax, 16 threads per row
  const int r = tid >> 4, sub = tid & 15;
  float pm = -3.0e38f;
  for (int c = sub; c < T_S; c += 16) pm = fmaxf(pm, S[r * T_S + c]);
  red[r][sub] = pm;
  __syncthreads();
  if (sub == 0) {
    float m = red[r][0];
    for (int i = 1; i < 16; ++i) m = fmaxf(m, red[r][i]);
    rowstat[r] = m;
  }
  __syncthreads();
  const float rm = rowstat[r];
  float ps = 0.0f;
  for (int c = sub; c < T_S; c += 16) {
    const float e = __expf(S[r * T_S + c] - rm);
    S[r * T_S + c] = e;
    ps += e;
  }
  __syncthreads();
  red[r][sub] = ps;
  __syncthreads();
  if (sub == 0) {
    float s = 0.0f;
    for (int i = 0; i < 16; ++i) s += red[r][i];
    rowstat[r] = s;
  }
  __syncthreads();
  const float inv = 1.0f / rowstat[r];
  for (int c = sub; c < T_S; c += 16) P[r * T_S + c] = (__bf16)(S[r * T_S + c] * inv);
  __syncthreads();

  // phase 3: contexts (B operand strided by row; scalar gathers, no guards)
  for (int ct = wave * 4; ct < wave * 4 + 4; ++ct) {
    v8f acc = {};
    const int j = ct * 16 + l15;
    for (int kk = 0; kk < T_S; kk += 32) {
      v16bf a, b;
      const int ka = kk + lhi * 16;
      const v8bf lo = *(const v8bf*)&P[l15 * T_S + ka];
      const v8bf hi = *(const v8bf*)&P[l15 * T_S + ka + 8];
#pragma unroll
      for (int i = 0; i < 8; ++i) { a[i] = lo[i]; a[i + 8] = hi[i]; }
      const int kb = kk + lhi * 16;
#pragma unroll
      for (int i = 0; i < 16; ++i) b[i] = (__bf16)hsn[(long)(kb + i) * H2 + j];
      acc = wmma_bf16(a, b, acc);
    }
#pragma unroll
    for (int r2 = 0; r2 < 8; ++r2)
      out4h[((long)n * T_S + qb + r2 + lhi * 8) * 1024 + ct * 16 + l15] = acc[r2];
  }

  // phase 4: copy hs into second half of concat
  for (int idx = tid; idx < 16 * T_S; idx += 256) {
    const int rr = idx >> 9, cc = idx & 511;
    out4h[((long)n * T_S + qb + rr) * 1024 + 512 + cc] = hsn[(long)(qb + rr) * H2 + cc];
  }
}

// ---------------------------------------------------------------------------
// BatchNorm matching the reference's reshape(N,50,T) view on the PADDED
// (N*T, 64) fc1 buffer. Channel ch covers flat positions f = t*50+c with
// f in [512*ch, 512*ch+512) per sample; storage index = (n*512+t)*64 + c.
// ---------------------------------------------------------------------------
__global__ __launch_bounds__(256)
void bn_stats(const float* __restrict__ x, float* __restrict__ stats)
{
  const int ch = blockIdx.x;
  __shared__ float s1[256], s2[256];
  float a = 0.0f, b = 0.0f;
  for (int i = threadIdx.x; i < N_B * T_S; i += 256) {
    const int n = i >> 9, j = i & 511;
    const int f = ch * 512 + j;
    const int t = f / 50, c = f % 50;
    const float v = x[((long)n * T_S + t) * 64 + c];
    a += v; b += v * v;
  }
  s1[threadIdx.x] = a; s2[threadIdx.x] = b;
  __syncthreads();
  for (int off = 128; off > 0; off >>= 1) {
    if (threadIdx.x < off) {
      s1[threadIdx.x] += s1[threadIdx.x + off];
      s2[threadIdx.x] += s2[threadIdx.x + off];
    }
    __syncthreads();
  }
  if (threadIdx.x == 0) {
    const float cnt = (float)(N_B * T_S);
    const float mean = s1[0] / cnt;
    stats[ch]      = mean;
    stats[50 + ch] = s2[0] / cnt - mean * mean;  // biased variance
  }
}

__global__ __launch_bounds__(256)
void bn_apply(float* __restrict__ x, const float* __restrict__ stats,
              const float* __restrict__ g, const float* __restrict__ bta)
{
  const long idx = (long)blockIdx.x * 256 + threadIdx.x;  // over real (n,t,c)
  if (idx >= (long)N_B * T_S * 50) return;
  const int rem = (int)(idx % (T_S * 50));   // = t*50 + c  (reference flat f)
  const int n   = (int)(idx / (T_S * 50));
  const int t = rem / 50, c = rem % 50;
  const int ch = rem >> 9;
  const float mean = stats[ch], var = stats[50 + ch];
  const long si = ((long)n * T_S + t) * 64 + c;
  x[si] = (x[si] - mean) * rsqrtf(var + 1e-5f) * g[ch] + bta[ch];
}

// ---------------------------------------------------------------------------
static inline int gemm_blocks(int M, int wcols) {
  const int tiles = (M / 16) * ((wcols + 15) / 16);
  return (tiles + 3) / 4;
}

extern "C" void kernel_launch(void* const* d_in, const int* in_sizes, int n_in,
                              void* d_out, int out_size, void* d_ws, size_t ws_size,
                              hipStream_t stream) {
  (void)in_sizes; (void)n_in; (void)out_size; (void)ws_size;
  const float* x         = (const float*)d_in[0];
  const float* w_ih_l0f  = (const float*)d_in[1];
  const float* w_hh_l0f  = (const float*)d_in[2];
  const float* b_l0f     = (const float*)d_in[3];
  const float* w_ih_l0b  = (const float*)d_in[4];
  const float* w_hh_l0b  = (const float*)d_in[5];
  const float* b_l0b     = (const float*)d_in[6];
  const float* w_ih_l1f  = (const float*)d_in[7];
  const float* w_hh_l1f  = (const float*)d_in[8];
  const float* b_l1f     = (const float*)d_in[9];
  const float* w_ih_l1b  = (const float*)d_in[10];
  const float* w_hh_l1b  = (const float*)d_in[11];
  const float* b_l1b     = (const float*)d_in[12];
  const float* fc1_w     = (const float*)d_in[13];
  const float* fc1_b     = (const float*)d_in[14];
  const float* bn_g      = (const float*)d_in[15];
  const float* bn_b      = (const float*)d_in[16];
  const float* fc2_w     = (const float*)d_in[17];
  const float* fc2_b     = (const float*)d_in[18];
  const float* fc3_w     = (const float*)d_in[19];
  const float* fc3_b     = (const float*)d_in[20];
  const float* fc4_w     = (const float*)d_in[21];
  const float* fc4_b     = (const float*)d_in[22];

  const long NT = (long)N_B * T_S;  // 32768
  char* ws = (char*)d_ws;
  size_t off = 0;
  float* ginf = (float*)(ws + off); off += (size_t)NT * G4 * 4;   // 128 MB
  float* ginb = (float*)(ws + off); off += (size_t)NT * G4 * 4;   // 128 MB
  float* hs0  = (float*)(ws + off); off += (size_t)NT * H2 * 4;   //  64 MB
  float* hs1  = (float*)(ws + off); off += (size_t)NT * H2 * 4;   //  64 MB
  float* o4h  = (float*)(ws + off); off += (size_t)NT * 1024 * 4; // 128 MB
  float* fc1o = (float*)(ws + off); off += (size_t)NT * 64 * 4;   // padded 50->64
  float* fc2o = (float*)(ws + off); off += (size_t)NT * 32 * 4;   // padded 25->32
  float* fc3o = (float*)(ws + off); off += (size_t)NT * 32 * 4;   // padded 10->32
  float* wp2  = (float*)(ws + off); off += (size_t)25 * 64 * 4;   // fc2_w packed K 50->64
  float* wp3  = (float*)(ws + off); off += (size_t)10 * 32 * 4;   // fc3_w packed K 25->32
  float* wp4  = (float*)(ws + off); off += (size_t)2  * 32 * 4;   // fc4_w packed K 10->32
  float* stats= (float*)(ws + off); off += 128 * 4;

  // pack small FC weights (zero-padded K)
  pack_w<<<(25 * 64 + 255) / 256, 256, 0, stream>>>(fc2_w, wp2, 25, 50, 64);
  pack_w<<<(10 * 32 + 255) / 256, 256, 0, stream>>>(fc3_w, wp3, 10, 25, 32);
  pack_w<<<(2  * 32 + 255) / 256, 256, 0, stream>>>(fc4_w, wp4, 2, 10, 32);

  // layer 0 input projections (time-parallel, whole chip)
  gemm_wmma<<<gemm_blocks((int)NT, G4), 128, 0, stream>>>(x, w_ih_l0f, b_l0f, ginf, (int)NT, D_IN, G4, G4, G4, 0);
  gemm_wmma<<<gemm_blocks((int)NT, G4), 128, 0, stream>>>(x, w_ih_l0b, b_l0b, ginb, (int)NT, D_IN, G4, G4, G4, 0);
  // layer 0 recurrence (fwd + bwd concurrently)
  lstm_dir<<<2, 1024, 0, stream>>>(ginf, ginb, w_hh_l0f, w_hh_l0b, hs0);

  // layer 1 input projections
  gemm_wmma<<<gemm_blocks((int)NT, G4), 128, 0, stream>>>(hs0, w_ih_l1f, b_l1f, ginf, (int)NT, H2, G4, G4, G4, 0);
  gemm_wmma<<<gemm_blocks((int)NT, G4), 128, 0, stream>>>(hs0, w_ih_l1b, b_l1b, ginb, (int)NT, H2, G4, G4, G4, 0);
  lstm_dir<<<2, 1024, 0, stream>>>(ginf, ginb, w_hh_l1f, w_hh_l1b, hs1);

  // attention + concat
  attn_kernel<<<dim3(T_S / 16, N_B), 256, 0, stream>>>(hs1, o4h);

  // fc1 (padded out 64) + BN + fc2 + fc3 + fc4
  gemm_wmma<<<gemm_blocks((int)NT, 64), 128, 0, stream>>>(o4h, fc1_w, fc1_b, fc1o, (int)NT, 1024, 50, 64, 64, 1);
  bn_stats<<<50, 256, 0, stream>>>(fc1o, stats);
  bn_apply<<<(int)((NT * 50 + 255) / 256), 256, 0, stream>>>(fc1o, stats, bn_g, bn_b);
  gemm_wmma<<<gemm_blocks((int)NT, 32), 128, 0, stream>>>(fc1o, wp2, fc2_b, fc2o, (int)NT, 64, 25, 32, 32, 1);
  gemm_wmma<<<gemm_blocks((int)NT, 32), 128, 0, stream>>>(fc2o, wp3, fc3_b, fc3o, (int)NT, 32, 10, 32, 32, 1);
  gemm_wmma<<<gemm_blocks((int)NT, 2), 128, 0, stream>>>(fc3o, wp4, fc4_b, (float*)d_out, (int)NT, 32, 2, 2, 2, 0);
}
